// FastSpeech2OneHotCondition_28484223107655
// MI455X (gfx1250) — matrix operand order repaired and checked by reading, not compiled, not run on hardware
//
#include <hip/hip_runtime.h>
#include <hip/hip_bf16.h>

// ---------------------------------------------------------------------------
// FastSpeech2-style forward for MI455X (gfx1250, wave32, WMMA).
// All GEMM/attention matmuls: v_wmma_f32_16x16x32_f16, operands fed by
// global_load_b128 of pre-converted f16 data (no per-element cvt in the loop).
// Flash attention: no score materialization; P staged through LDS for the
// C-layout -> A-layout transpose.
// ---------------------------------------------------------------------------

typedef __attribute__((ext_vector_type(16))) _Float16 v16h;
typedef __attribute__((ext_vector_type(8)))  _Float16 v8h;
typedef __attribute__((ext_vector_type(8)))  float    v8f;

union V16 { v16h v; v8h h[2]; };

#define BSZ     8
#define T_TEXT  128
#define T_MEL   1024
#define DMODEL  384
#define NHEAD   6
#define DHEAD   64
#define CONDD   64
#define NMELS   80
#define DFF     768

__device__ __forceinline__ v8f wmma16(v16h a, v16h b, v8f c) {
  // (neg_a, A, neg_b, B, c_mod, C, reuse_a, reuse_b)
  return __builtin_amdgcn_wmma_f32_16x16x32_f16(false, a, false, b, (short)0, c,
                                                false, false);
}

// A-matrix (16x32 f16) K-offset for VGPR pair t, lane half (ISA 7.12.2).
__device__ __forceinline__ int a_koff(int t, int half) {
  return ((t < 4) ? (2 * t) : (16 + 2 * (t - 4))) + 8 * half;
}

// ---------------------------------------------------------------------------
// WMMA GEMM, register-blocked 16x(16*NT) per wave.
//   C[M,N] = act(A[M,K]_f16 @ W[K,N] + bias (+ addend_f32))
// W supplied pre-transposed f16: WT[N,K]  -> B operand = 2x b128 loads.
// A operand = 2x b128 loads (lane pattern 8*half + 0..7 / 16+8*half + 0..7).
// Output: f32 (Cf) or f16 (Ch); trans!=0 writes Ch transposed [N,M] (for V^T).
// ---------------------------------------------------------------------------
template <int NT>
__global__ __launch_bounds__(256)
void k_gemm_wmma(const _Float16* __restrict__ A, const _Float16* __restrict__ WT,
                 const float* __restrict__ bias, const float* __restrict__ addend,
                 float* __restrict__ Cf, _Float16* __restrict__ Ch,
                 int M, int N, int K, int relu, int trans)
{
  int wave = (blockIdx.x * blockDim.x + threadIdx.x) >> 5;
  int gN   = N / (16 * NT);
  int tiles = (M >> 4) * gN;
  if (wave >= tiles) return;                 // wave-uniform: EXEC all-1s
  int mt = wave / gN, ng = wave % gN;
  int lane = threadIdx.x & 31;
  int half = lane >> 4, l16 = lane & 15;

  const _Float16* Arow = A + (size_t)(mt * 16 + l16) * K;
  const _Float16* Brow[NT];
#pragma unroll
  for (int s = 0; s < NT; ++s)
    Brow[s] = WT + (size_t)((ng * NT + s) * 16 + l16) * K;

  v8f acc[NT];
#pragma unroll
  for (int s = 0; s < NT; ++s) acc[s] = (v8f){};

  for (int k0 = 0; k0 < K; k0 += 32) {
    V16 av;                                  // A: k = k0 + a_koff(t, half)
    av.h[0] = *(const v8h*)(Arow + k0 + 8 * half);
    av.h[1] = *(const v8h*)(Arow + k0 + 16 + 8 * half);
#pragma unroll
    for (int s = 0; s < NT; ++s) {
      V16 bv;                                // B: k = k0 + a_koff(t, half)
      bv.h[0] = *(const v8h*)(Brow[s] + k0 + 8 * half);
      bv.h[1] = *(const v8h*)(Brow[s] + k0 + 16 + 8 * half);
      acc[s] = wmma16(av.v, bv.v, acc[s]);
    }
  }

#pragma unroll
  for (int s = 0; s < NT; ++s) {
    int col = (ng * NT + s) * 16 + l16;
    float bb = bias ? bias[col] : 0.f;
    if (trans) {                             // packed 16B store of a column
      v8h ph;
#pragma unroll
      for (int r = 0; r < 8; ++r) ph[r] = (_Float16)(acc[s][r] + bb);
      *(v8h*)(Ch + (size_t)col * M + mt * 16 + 8 * half) = ph;
    } else {
#pragma unroll
      for (int r = 0; r < 8; ++r) {          // C/D row = r + 8*half
        int row = mt * 16 + r + 8 * half;
        float v = acc[s][r] + bb;
        if (addend) v += addend[(size_t)row * N + col];
        if (relu)   v = fmaxf(v, 0.f);
        if (Ch) Ch[(size_t)row * N + col] = (_Float16)v;
        else    Cf[(size_t)row * N + col] = v;
      }
    }
  }
}

// ---------------------------------------------------------------------------
// Fused flash attention: one wave per (b, h, 16-row Q tile).
// Q,K f16 row-major [Mtot, 384]; V^T f16 [384, Mtot]; O f16 [Mtot, 384].
// All WMMA operands are contiguous b128 loads. Softmax scale applied to f32 S.
// ---------------------------------------------------------------------------
__global__ __launch_bounds__(256)
void k_attn_flash(const _Float16* __restrict__ Q, const _Float16* __restrict__ K,
                  const _Float16* __restrict__ VT, _Float16* __restrict__ O,
                  int T, int Mtot)
{
  __shared__ alignas(16) _Float16 Plds[8][16 * 32];   // per-wave P staging

  int wid    = threadIdx.x >> 5;
  int wave   = blockIdx.x * (blockDim.x >> 5) + wid;
  int qtiles = T >> 4;
  int total  = BSZ * NHEAD * qtiles;
  if (wave >= total) return;                 // wave-uniform
  int qt = wave % qtiles;
  int bh = wave / qtiles;
  int h  = bh % NHEAD, b = bh / NHEAD;

  int lane = threadIdx.x & 31;
  int half = lane >> 4, l16 = lane & 15;
  const int Dm = NHEAD * DHEAD;              // 384
  const float scale = 0.125f;                // 1/sqrt(64)

  const _Float16* Qb = Q + (size_t)(b * T) * Dm + h * DHEAD;
  const _Float16* Kb = K + (size_t)(b * T) * Dm + h * DHEAD;

  // Q tile (16x64) in A layout; qa[kk] covers dh = kk*32..+31
  V16 qa[2];
  {
    const _Float16* qrow = Qb + (size_t)(qt * 16 + l16) * Dm;
#pragma unroll
    for (int kk = 0; kk < 2; ++kk) {
      qa[kk].h[0] = *(const v8h*)(qrow + kk * 32 + 8 * half);
      qa[kk].h[1] = *(const v8h*)(qrow + kk * 32 + 16 + 8 * half);
    }
  }

  float mrun[8], lrun[8];
  v8f oacc[4];
#pragma unroll
  for (int r = 0; r < 8; ++r) { mrun[r] = -1e30f; lrun[r] = 0.f; }
#pragma unroll
  for (int nt = 0; nt < 4; ++nt) oacc[nt] = (v8f){};

  _Float16* P = &Plds[wid][0];

  for (int j0 = 0; j0 < T; j0 += 32) {       // 32 keys per iteration
    // S chunk: two 16x16 C tiles (keys j0+s*16 .. +15)
    v8f schunk[2];
#pragma unroll
    for (int s = 0; s < 2; ++s) {
      v8f sc = {};
      const _Float16* krow = Kb + (size_t)(j0 + s * 16 + l16) * Dm;
#pragma unroll
      for (int kk = 0; kk < 2; ++kk) {       // reduce dh in 32-steps
        V16 bv;                              // B: k = kk*32 + a_koff(t, half)
        bv.h[0] = *(const v8h*)(krow + kk * 32 + 8 * half);
        bv.h[1] = *(const v8h*)(krow + kk * 32 + 16 + 8 * half);
        sc = wmma16(qa[kk].v, bv.v, sc);
      }
      schunk[s] = sc;
    }

    // Online softmax per owned row (row = r + 8*half); 16-lane reductions
    float p0[8], p1[8];
#pragma unroll
    for (int r = 0; r < 8; ++r) {
      float v0 = schunk[0][r] * scale, v1 = schunk[1][r] * scale;
      float mx = fmaxf(v0, v1);
#pragma unroll
      for (int off = 8; off >= 1; off >>= 1)
        mx = fmaxf(mx, __shfl_xor(mx, off, 16));
      float mnew = fmaxf(mrun[r], mx);
      float corr = __expf(mrun[r] - mnew);
      float e0 = __expf(v0 - mnew), e1 = __expf(v1 - mnew);
      float rs = e0 + e1;
#pragma unroll
      for (int off = 8; off >= 1; off >>= 1)
        rs += __shfl_xor(rs, off, 16);
      lrun[r] = lrun[r] * corr + rs;
      mrun[r] = mnew;
#pragma unroll
      for (int nt = 0; nt < 4; ++nt) oacc[nt][r] *= corr;
      p0[r] = e0; p1[r] = e1;
    }

    // Stage P (16x32 f16 row-major) in LDS; wave-internal DS ops are in-order
#pragma unroll
    for (int r = 0; r < 8; ++r) {
      int row = r + 8 * half;
      P[row * 32 + l16]      = (_Float16)p0[r];
      P[row * 32 + 16 + l16] = (_Float16)p1[r];
    }
    // Re-read P in A layout: two aligned 16B LDS reads
    V16 pa;
    pa.h[0] = *(const v8h*)(P + l16 * 32 + 8 * half);
    pa.h[1] = *(const v8h*)(P + l16 * 32 + 16 + 8 * half);

    // O += P @ V[j0:j0+32, 0:64] via V^T (contiguous keys -> b128 loads)
#pragma unroll
    for (int nt = 0; nt < 4; ++nt) {
      int dcol = h * DHEAD + nt * 16 + l16;
      const _Float16* vrow = VT + (size_t)dcol * Mtot + b * T + j0;
      V16 vb;                                // B: k(=key) = a_koff(t, half)
      vb.h[0] = *(const v8h*)(vrow + 8 * half);
      vb.h[1] = *(const v8h*)(vrow + 16 + 8 * half);
      oacc[nt] = wmma16(pa.v, vb.v, oacc[nt]);
    }
  }

  // Normalize and write O (f16, feeds the Wo WMMA GEMM)
  _Float16* Ob = O + (size_t)(b * T) * Dm + h * DHEAD;
#pragma unroll
  for (int r = 0; r < 8; ++r) {
    int row = qt * 16 + r + 8 * half;
    float inv = 1.f / lrun[r];
#pragma unroll
    for (int nt = 0; nt < 4; ++nt)
      Ob[(size_t)row * Dm + nt * 16 + l16] = (_Float16)(oacc[nt][r] * inv);
  }
}

// ---------------------------------------------------------------------------
// LayerNorm over D=384: one wave per row; f32 in, f16 out (feeds WMMA A).
// ---------------------------------------------------------------------------
__global__ __launch_bounds__(256)
void k_layernorm(const float* __restrict__ x, const float* __restrict__ g,
                 const float* __restrict__ bta, _Float16* __restrict__ out, int M)
{
  int wave = (blockIdx.x * blockDim.x + threadIdx.x) >> 5;
  if (wave >= M) return;
  int lane = threadIdx.x & 31;
  const float* row = x + (size_t)wave * DMODEL;
  float xs[12], s = 0.f;
#pragma unroll
  for (int i = 0; i < 12; ++i) { xs[i] = row[lane + 32 * i]; s += xs[i]; }
#pragma unroll
  for (int off = 16; off >= 1; off >>= 1) s += __shfl_xor(s, off, 32);
  float mean = s * (1.f / DMODEL);
  float v = 0.f;
#pragma unroll
  for (int i = 0; i < 12; ++i) { float d = xs[i] - mean; v += d * d; }
#pragma unroll
  for (int off = 16; off >= 1; off >>= 1) v += __shfl_xor(v, off, 32);
  float inv = rsqrtf(v * (1.f / DMODEL) + 1e-5f);
  _Float16* orow = out + (size_t)wave * DMODEL;
#pragma unroll
  for (int i = 0; i < 12; ++i) {
    int d = lane + 32 * i;
    orow[d] = (_Float16)((xs[i] - mean) * inv * g[d] + bta[d]);
  }
}

// ---------------------------------------------------------------------------
// One-time converters
// ---------------------------------------------------------------------------
// WT[n*K+k] = (f16) W[k*N+n]   (weight transpose+convert, runs once per call)
__global__ void k_w2h_t(const float* __restrict__ W, _Float16* __restrict__ WT,
                        int K, int N)
{
  int t = blockIdx.x * blockDim.x + threadIdx.x;
  if (t >= K * N) return;
  int n = t / K, k = t % K;
  WT[t] = (_Float16)W[(size_t)k * N + n];
}

__global__ void k_f2h(const float* __restrict__ in, _Float16* __restrict__ out, int n)
{
  int t = blockIdx.x * blockDim.x + threadIdx.x;
  if (t < n) out[t] = (_Float16)in[t];
}

// ---------------------------------------------------------------------------
// Small pipeline kernels
// ---------------------------------------------------------------------------
__global__ void k_embed(const int* __restrict__ tok, const float* __restrict__ emb,
                        float* __restrict__ x)
{
  int t = blockIdx.x * blockDim.x + threadIdx.x;
  if (t >= BSZ * T_TEXT * DMODEL) return;
  int d = t % DMODEL, m = t / DMODEL;
  x[t] = emb[tok[m] * DMODEL + d];
}

__global__ void k_concat_cond(const float* __restrict__ x, const float* __restrict__ emo,
                              const int* __restrict__ eidx, _Float16* __restrict__ out)
{
  const int CW = DMODEL + CONDD;             // 448
  int t = blockIdx.x * blockDim.x + threadIdx.x;
  if (t >= BSZ * T_TEXT * CW) return;
  int c = t % CW, m = t / CW, b = m / T_TEXT;
  float v = (c < DMODEL) ? x[m * DMODEL + c]
                         : emo[eidx[b] * CONDD + (c - DMODEL)];
  out[t] = (_Float16)v;
}

// f0 / energy / dur(softplus) heads (N=1 dots): one wave per row, f16 input.
__global__ __launch_bounds__(256)
void k_heads(const _Float16* __restrict__ h,
             const float* f0W, const float* f0b, const float* enW, const float* enb,
             const float* duW, const float* dub,
             float* __restrict__ of0, float* __restrict__ oen,
             float* __restrict__ odu, int M)
{
  int wave = (blockIdx.x * blockDim.x + threadIdx.x) >> 5;
  if (wave >= M) return;
  int lane = threadIdx.x & 31;
  const _Float16* row = h + (size_t)wave * DMODEL;
  float a = 0.f, b = 0.f, c = 0.f;
#pragma unroll
  for (int i = 0; i < 12; ++i) {
    int d = lane + 32 * i;
    float x = (float)row[d];
    a += x * f0W[d]; b += x * enW[d]; c += x * duW[d];
  }
#pragma unroll
  for (int off = 16; off >= 1; off >>= 1) {
    a += __shfl_xor(a, off, 32);
    b += __shfl_xor(b, off, 32);
    c += __shfl_xor(c, off, 32);
  }
  if (lane == 0) {
    of0[wave] = a + f0b[0];
    oen[wave] = b + enb[0];
    float z = c + dub[0];
    odu[wave] = (z > 20.f) ? z : log1pf(__expf(z));   // softplus
  }
}

__global__ void k_cumsum_dur(const int* __restrict__ dur, int* __restrict__ ends)
{
  int b = blockIdx.x * blockDim.x + threadIdx.x;
  if (b >= BSZ) return;
  int s = 0;
  for (int t = 0; t < T_TEXT; ++t) {
    int d = dur[b * T_TEXT + t];
    if (d < 0) d = 0;
    s += d;
    ends[b * T_TEXT + t] = s;
  }
}

// searchsorted(ends, frame, side='right'); -1 marks invalid frames.
__global__ void k_lr_index(const int* __restrict__ ends, int* __restrict__ idxv)
{
  int m = blockIdx.x * blockDim.x + threadIdx.x;
  if (m >= BSZ * T_MEL) return;
  int b = m / T_MEL, f = m % T_MEL;
  const int* e = ends + b * T_TEXT;
  int lo = 0, hi = T_TEXT;
  while (lo < hi) { int mid = (lo + hi) >> 1; if (e[mid] <= f) lo = mid + 1; else hi = mid; }
  int idx = (lo < T_TEXT) ? lo : (T_TEXT - 1);
  idxv[m] = (f < e[T_TEXT - 1]) ? idx : -1;
}

// c_emb[b,d] = cond[b] @ mel_proj_W + mel_proj_b  (tiny: 8x384, K=64)
__global__ void k_cemb(const float* __restrict__ emo, const int* __restrict__ eidx,
                       const float* __restrict__ mW, const float* __restrict__ mb,
                       float* __restrict__ cemb)
{
  int t = blockIdx.x * blockDim.x + threadIdx.x;
  if (t >= BSZ * DMODEL) return;
  int b = t / DMODEL, d = t % DMODEL;
  const float* c = emo + eidx[b] * CONDD;
  float s = mb[d];
  for (int k = 0; k < CONDD; ++k) s += c[k] * mW[k * DMODEL + d];
  cemb[t] = s;
}

// y[b,f,:] = (valid ? x_enc[b,idx,:] : 0) + c_emb[b,:]
__global__ void k_gather_add(const float* __restrict__ xenc, const int* __restrict__ idxv,
                             const float* __restrict__ cemb, float* __restrict__ y)
{
  int t = blockIdx.x * blockDim.x + threadIdx.x;
  if (t >= BSZ * T_MEL * DMODEL) return;
  int d = t % DMODEL, m = t / DMODEL, b = m / T_MEL;
  int iv = idxv[m];
  float v = (iv >= 0) ? xenc[((size_t)b * T_TEXT + iv) * DMODEL + d] : 0.f;
  y[t] = v + cemb[b * DMODEL + d];
}

// ---------------------------------------------------------------------------
// Host-side orchestration
// ---------------------------------------------------------------------------
extern "C" void kernel_launch(void* const* d_in, const int* in_sizes, int n_in,
                              void* d_out, int out_size, void* d_ws, size_t ws_size,
                              hipStream_t stream)
{
  (void)in_sizes; (void)n_in; (void)out_size; (void)ws_size;

  const int* tokens   = (const int*)d_in[0];
  const int* eidx     = (const int*)d_in[1];
  const int* duration = (const int*)d_in[2];
  const float* tok_emb = (const float*)d_in[3];
  const float* emo_emb = (const float*)d_in[4];
  auto P = [&](int i) { return (const float*)d_in[i]; };
  // Layer block of 16: +0 ln1_g +1 ln1_b +2 Wq +3 bq +4 Wk +5 bk +6 Wv +7 bv
  // +8 Wo +9 bo +10 ln2_g +11 ln2_b +12 W1 +13 b1 +14 W2 +15 b2.
  // enc layers @5,21; dec layers @37,53; 69 va_W 70 va_b 71..76 heads,
  // 77 mel_proj_W 78 mel_proj_b 79 out_W 80 out_b.

  // --- workspace carve-up (all segments 256B aligned) ---
  char* base = (char*)d_ws;
  size_t off = 0;
  auto alloc = [&](size_t bytes) {
    void* p = base + off;
    off += (bytes + 255) & ~(size_t)255;
    return p;
  };
  const size_t SEG = (size_t)BSZ * T_MEL * DMODEL;     // 3,145,728 elems
  float*    b0   = (float*)alloc(SEG * 4);             // enc x | dec scratch
  float*    b1   = (float*)alloc(SEG * 4);             // scratch | dec residual
  _Float16* hn   = (_Float16*)alloc(SEG * 2);          // LN out / attn O (f16)
  _Float16* qh   = (_Float16*)alloc(SEG * 2);          // Q (f16)
  _Float16* kh   = (_Float16*)alloc(SEG * 2);          // K (f16); FF hidden spans kh+vt
  _Float16* vt   = (_Float16*)alloc(SEG * 2);          // V^T (f16)
  float*    cemb = (float*)alloc(BSZ * DMODEL * 4);
  int*      ends = (int*)alloc(BSZ * T_TEXT * 4);
  int*      idxv = (int*)alloc(BSZ * T_MEL * 4);

  // f16 transposed weight pool
  auto wt_alloc = [&](size_t elems) { return (_Float16*)alloc(elems * 2); };
  _Float16 *wq_t[4], *wk_t[4], *wv_t[4], *wo_t[4], *w1_t[4], *w2_t[4];
  int lbase[4] = {5, 21, 37, 53};
  for (int l = 0; l < 4; ++l) {
    wq_t[l] = wt_alloc((size_t)DMODEL * DMODEL);
    wk_t[l] = wt_alloc((size_t)DMODEL * DMODEL);
    wv_t[l] = wt_alloc((size_t)DMODEL * DMODEL);
    wo_t[l] = wt_alloc((size_t)DMODEL * DMODEL);
    w1_t[l] = wt_alloc((size_t)DMODEL * DFF);
    w2_t[l] = wt_alloc((size_t)DFF * DMODEL);
  }
  _Float16* va_t  = wt_alloc((size_t)(DMODEL + CONDD) * DMODEL);
  _Float16* out_t = wt_alloc((size_t)DMODEL * NMELS);

  float* out_mel = (float*)d_out;                          // [8*1024, 80]
  float* out_f0  = out_mel + (size_t)BSZ * T_MEL * NMELS;  // [1024]
  float* out_en  = out_f0 + BSZ * T_TEXT;
  float* out_du  = out_en + BSZ * T_TEXT;

  auto w2h = [&](const float* W, _Float16* WT, int K, int N) {
    int n = K * N;
    k_w2h_t<<<(n + 255) / 256, 256, 0, stream>>>(W, WT, K, N);
  };
  // convert + transpose every WMMA weight once
  for (int l = 0; l < 4; ++l) {
    int bI = lbase[l];
    w2h(P(bI + 2),  wq_t[l], DMODEL, DMODEL);
    w2h(P(bI + 4),  wk_t[l], DMODEL, DMODEL);
    w2h(P(bI + 6),  wv_t[l], DMODEL, DMODEL);
    w2h(P(bI + 8),  wo_t[l], DMODEL, DMODEL);
    w2h(P(bI + 12), w1_t[l], DMODEL, DFF);
    w2h(P(bI + 14), w2_t[l], DFF, DMODEL);
  }
  w2h(P(69), va_t, DMODEL + CONDD, DMODEL);
  w2h(P(79), out_t, DMODEL, NMELS);

  auto gemm4 = [&](const _Float16* A, const _Float16* WT, const float* bias,
                   const float* add, float* Cf, _Float16* Ch,
                   int M, int N, int K, int relu, int trans) {
    int tiles = (M / 16) * (N / 64);
    k_gemm_wmma<4><<<(tiles + 7) / 8, 256, 0, stream>>>(
        A, WT, bias, add, Cf, Ch, M, N, K, relu, trans);
  };
  auto lnorm = [&](const float* x, const float* g, const float* b,
                   _Float16* o, int M) {
    k_layernorm<<<(M + 7) / 8, 256, 0, stream>>>(x, g, b, o, M);
  };

  auto xf_layer = [&](float* x, int l, int M, int T) {
    int bI = lbase[l];
    lnorm(x, P(bI + 0), P(bI + 1), hn, M);
    gemm4(hn, wq_t[l], P(bI + 3), nullptr, nullptr, qh, M, DMODEL, DMODEL, 0, 0);
    gemm4(hn, wk_t[l], P(bI + 5), nullptr, nullptr, kh, M, DMODEL, DMODEL, 0, 0);
    gemm4(hn, wv_t[l], P(bI + 7), nullptr, nullptr, vt, M, DMODEL, DMODEL, 0, 1); // V^T
    int waves = BSZ * NHEAD * (T / 16);
    k_attn_flash<<<(waves + 7) / 8, 256, 0, stream>>>(qh, kh, vt, hn, T, M);
    gemm4(hn, wo_t[l], P(bI + 9), x, x, nullptr, M, DMODEL, DMODEL, 0, 0);  // x += O@Wo
    lnorm(x, P(bI + 10), P(bI + 11), hn, M);
    gemm4(hn, w1_t[l], P(bI + 13), nullptr, nullptr, kh, M, DFF, DMODEL, 1, 0); // relu FF1
    gemm4(kh, w2_t[l], P(bI + 15), x, x, nullptr, M, DMODEL, DFF, 0, 0);    // x += FF2
  };

  // ---- Encoder (M = 1024, T = 128) ----
  {
    int n = BSZ * T_TEXT * DMODEL;
    k_embed<<<(n + 255) / 256, 256, 0, stream>>>(tokens, tok_emb, b0);
  }
  xf_layer(b0, 0, BSZ * T_TEXT, T_TEXT);
  xf_layer(b0, 1, BSZ * T_TEXT, T_TEXT);

  // ---- Variance adaptor ----
  {
    int n = BSZ * T_TEXT * (DMODEL + CONDD);
    k_concat_cond<<<(n + 255) / 256, 256, 0, stream>>>(b0, emo_emb, eidx, hn);
  }
  gemm4(hn, va_t, P(70), nullptr, nullptr, qh,
        BSZ * T_TEXT, DMODEL, DMODEL + CONDD, 1, 0);
  k_heads<<<(BSZ * T_TEXT + 7) / 8, 256, 0, stream>>>(
      qh, P(71), P(72), P(73), P(74), P(75), P(76),
      out_f0, out_en, out_du, BSZ * T_TEXT);

  // ---- Length regulate + conditioning ----
  k_cumsum_dur<<<1, 32, 0, stream>>>(duration, ends);
  {
    int n = BSZ * T_MEL;
    k_lr_index<<<(n + 255) / 256, 256, 0, stream>>>(ends, idxv);
  }
  {
    int n = BSZ * DMODEL;
    k_cemb<<<(n + 255) / 256, 256, 0, stream>>>(emo_emb, eidx, P(77), P(78), cemb);
  }
  {
    int n = BSZ * T_MEL * DMODEL;
    k_gather_add<<<(n + 255) / 256, 256, 0, stream>>>(b0, idxv, cemb, b1);
  }

  // ---- Decoder (M = 8192, T = 1024) ----
  xf_layer(b1, 2, BSZ * T_MEL, T_MEL);
  xf_layer(b1, 3, BSZ * T_MEL, T_MEL);

  // ---- Mel projection: convert residual to f16, then WMMA into d_out ----
  {
    int n = (int)SEG;
    k_f2h<<<(n + 255) / 256, 256, 0, stream>>>(b1, hn, n);
  }
  {
    int tiles = (BSZ * T_MEL / 16) * (NMELS / 80);     // NT=5 -> one 16x80 strip
    k_gemm_wmma<5><<<(tiles + 7) / 8, 256, 0, stream>>>(
        hn, out_t, P(80), nullptr, out_mel, nullptr,
        BSZ * T_MEL, NMELS, DMODEL, 0, 0);
  }
}
